// SparseAutoencoder_31885837205915
// MI455X (gfx1250) — matrix operand order repaired and checked by reading, not compiled, not run on hardware
//
#include <hip/hip_runtime.h>
#include <hip/hip_bf16.h>

#define BDIM 16384
#define DDIM 768
#define HDIM 12288
#define KSP  32
#define MAXACT 256

typedef __attribute__((ext_vector_type(16))) __bf16   v16bf;
typedef __attribute__((ext_vector_type(8)))  float    v8f;
typedef __attribute__((ext_vector_type(4)))  unsigned u32x4;

union FragB16 { v16bf v; u32x4 q[2]; };

__device__ __forceinline__ unsigned short f32_to_bf16(float f) {
    unsigned u = __float_as_uint(f);
    unsigned rnd = 0x7FFFu + ((u >> 16) & 1u);   // round-to-nearest-even
    u += rnd;
    return (unsigned short)(u >> 16);
}

__device__ __forceinline__ unsigned sortkey(float f) {
    unsigned u = __float_as_uint(f);
    return u ^ (((unsigned)((int)u >> 31)) | 0x80000000u);  // monotonic: bigger float -> bigger key
}

// ---------------------------------------------------------------- kernel 1: x -> bf16
__global__ void cvt_x_kernel(const float* __restrict__ x, unsigned short* __restrict__ xb, int n) {
    int i = blockIdx.x * blockDim.x + threadIdx.x;
    if (i < n) xb[i] = f32_to_bf16(x[i]);
}

// ---------------------------------------------------------------- kernel 2: W -> bf16 + row inv-norms
__global__ void prep_w_kernel(const float* __restrict__ W, unsigned short* __restrict__ wb,
                              float* __restrict__ invn) {
    __shared__ float red[256];
    int row = blockIdx.x;
    int tid = threadIdx.x;
    float s = 0.f;
    const float* wr = W + (size_t)row * DDIM;
    unsigned short* wo = wb + (size_t)row * DDIM;
    for (int c = tid; c < DDIM; c += 256) {
        float v = wr[c];
        wo[c] = f32_to_bf16(v);
        s += v * v;
    }
    red[tid] = s; __syncthreads();
    for (int off = 128; off > 0; off >>= 1) {
        if (tid < off) red[tid] += red[tid + off];
        __syncthreads();
    }
    if (tid == 0) invn[row] = rsqrtf(red[0] + 1e-24f);
}

// ---------------------------------------------------------------- kernel 3: pre = x * W^T + b  (bf16 WMMA)
// block = 256 threads = 8 waves arranged 2(M) x 4(N); block tile 128(M) x 256(N);
// wave tile 64x64 = 4x4 grid of 16x16x32 bf16 WMMAs (128 acc VGPRs, ~32 FLOP per L2 byte)
__global__ void __launch_bounds__(256)
gemm_pre_kernel(const unsigned short* __restrict__ xb,   // [B][D] bf16
                const unsigned short* __restrict__ wb,   // [H][D] bf16
                const float* __restrict__ bias,          // [H]
                float* __restrict__ pre)                 // [B][H]
{
    const int lane  = threadIdx.x & 31;
    const int wave  = threadIdx.x >> 5;
    const int half  = lane >> 4;      // 0/1
    const int r     = lane & 15;
    const int mwave = wave >> 2;      // 0..1
    const int nwave = wave & 3;       // 0..3

    const int m0 = blockIdx.y * 128 + mwave * 64;   // wave's row base
    const int n0 = blockIdx.x * 256 + nwave * 64;   // wave's col base

    const v8f zero = {0.f,0.f,0.f,0.f,0.f,0.f,0.f,0.f};
    v8f acc[4][4];
    #pragma unroll
    for (int mi = 0; mi < 4; ++mi)
        #pragma unroll
        for (int ni = 0; ni < 4; ++ni) acc[mi][ni] = zero;

    for (int k0 = 0; k0 < DDIM; k0 += 32) {
        FragB16 a[4], bf[4];
        // A (16x32 bf16): lane<16 row=r holds K {0..7,16..23}; lane>=16 holds K {8..15,24..31}
        #pragma unroll
        for (int mi = 0; mi < 4; ++mi) {
            const unsigned short* ap = xb + (size_t)(m0 + 16*mi + r) * DDIM + k0 + half * 8;
            a[mi].q[0] = *(const u32x4*)(ap);
            a[mi].q[1] = *(const u32x4*)(ap + 16);
        }
        // B (32x16 bf16): col = r, lane<16 holds K {0..7,16..23}; lane>=16 holds K {8..15,24..31}
        #pragma unroll
        for (int ni = 0; ni < 4; ++ni) {
            const unsigned short* bp = wb + (size_t)(n0 + 16*ni + r) * DDIM + k0 + half * 8;
            bf[ni].q[0] = *(const u32x4*)(bp);
            bf[ni].q[1] = *(const u32x4*)(bp + 16);
        }
        if (k0 + 32 < DDIM) {   // near-cache prefetch of next K slab
            __builtin_prefetch(xb + (size_t)(m0 + r) * DDIM + k0 + 32, 0, 3);
            __builtin_prefetch(wb + (size_t)(n0 + r) * DDIM + k0 + 32, 0, 3);
        }
        #pragma unroll
        for (int mi = 0; mi < 4; ++mi)
            #pragma unroll
            for (int ni = 0; ni < 4; ++ni)
                acc[mi][ni] = __builtin_amdgcn_wmma_f32_16x16x32_bf16(
                    false, a[mi].v, false, bf[ni].v, (short)0, acc[mi][ni], false, false);
    }

    // C/D layout: vgpr v, lane -> row M = v + 8*half, col N = r
    #pragma unroll
    for (int mi = 0; mi < 4; ++mi) {
        #pragma unroll
        for (int ni = 0; ni < 4; ++ni) {
            const int col = n0 + 16*ni + r;
            const float bb = bias[col];
            #pragma unroll
            for (int v = 0; v < 8; ++v) {
                const int row = m0 + 16*mi + v + 8*half;
                pre[(size_t)row * HDIM + col] = acc[mi][ni][v] + bb;
            }
        }
    }
}

// ---------------------------------------------------------------- kernel 4: per-row top-K radix select + sparse decode
__global__ void __launch_bounds__(256)
topk_decode_kernel(const float* __restrict__ pre,   // [B][H]
                   const float* __restrict__ W,     // [H][D] f32
                   const float* __restrict__ invn,  // [H]
                   float* __restrict__ out)         // [B][D]
{
    __shared__ float   rowv[HDIM];      // 48 KB
    __shared__ unsigned hist[256];
    __shared__ unsigned s_prefix, s_need;
    __shared__ int      s_nact;
    __shared__ int      act_i[MAXACT];
    __shared__ float    act_v[MAXACT];

    const int row = blockIdx.x;
    const int tid = threadIdx.x;
    const float* prow = pre + (size_t)row * HDIM;

    for (int i = tid; i < HDIM; i += 256) rowv[i] = prow[i];
    if (tid == 0) { s_prefix = 0u; s_need = KSP; s_nact = 0; }
    __syncthreads();

    // exact radix-select of the K-th largest key, 4 rounds of 8 bits (MSB first)
    unsigned prefixMask = 0u;
    for (int shift = 24; shift >= 0; shift -= 8) {
        for (int i = tid; i < 256; i += 256) hist[i] = 0u;
        __syncthreads();
        const unsigned pfx = s_prefix;
        for (int i = tid; i < HDIM; i += 256) {
            unsigned u = sortkey(rowv[i]);
            if ((u & prefixMask) == pfx)
                atomicAdd(&hist[(u >> shift) & 255u], 1u);
        }
        __syncthreads();
        if (tid == 0) {
            unsigned need = s_need, cum = 0u;
            for (int d = 255; d >= 0; --d) {
                unsigned c = hist[d];
                if (cum + c >= need) {
                    s_prefix = pfx | ((unsigned)d << shift);
                    s_need   = need - cum;
                    break;
                }
                cum += c;
            }
        }
        prefixMask |= (255u << shift);
        __syncthreads();
    }

    // invert key -> threshold value; encoded = pre * (pre >= thresh)
    const unsigned tk = s_prefix;
    const unsigned ub = (tk & 0x80000000u) ? (tk ^ 0x80000000u) : ~tk;
    const float thresh = __uint_as_float(ub);

    for (int i = tid; i < HDIM; i += 256) {
        float v = rowv[i];
        if (v >= thresh) {
            int pos = atomicAdd(&s_nact, 1);
            if (pos < MAXACT) { act_i[pos] = i; act_v[pos] = v; }
        }
    }
    __syncthreads();
    const int nact = (s_nact < MAXACT) ? s_nact : MAXACT;

    // fold in row inverse-norms once
    if (tid < nact) act_v[tid] *= invn[act_i[tid]];
    __syncthreads();

    // out[row, d] = sum_j act_v[j] * W[act_i[j], d]   (W is L2-resident)
    for (int d = tid; d < DDIM; d += 256) {
        float acc = 0.f;
        for (int j = 0; j < nact; ++j)
            acc += act_v[j] * W[(size_t)act_i[j] * DDIM + d];
        out[(size_t)row * DDIM + d] = acc;
    }
}

// ---------------------------------------------------------------- launcher
extern "C" void kernel_launch(void* const* d_in, const int* in_sizes, int n_in,
                              void* d_out, int out_size, void* d_ws, size_t ws_size,
                              hipStream_t stream) {
    const float* x = (const float*)d_in[0];   // [B][D]
    const float* W = (const float*)d_in[1];   // [H][D]
    const float* b = (const float*)d_in[2];   // [H]
    float* out = (float*)d_out;               // [B][D]

    char* ws = (char*)d_ws;
    size_t off = 0;
    unsigned short* xb = (unsigned short*)(ws + off); off += (size_t)BDIM * DDIM * sizeof(unsigned short);
    unsigned short* wb = (unsigned short*)(ws + off); off += (size_t)HDIM * DDIM * sizeof(unsigned short);
    float* invn        = (float*)(ws + off);          off += (size_t)HDIM * sizeof(float);
    off = (off + 255) & ~(size_t)255;
    float* pre         = (float*)(ws + off);          off += (size_t)BDIM * HDIM * sizeof(float);

    // 1) x -> bf16
    {
        int n = BDIM * DDIM;
        cvt_x_kernel<<<(n + 255) / 256, 256, 0, stream>>>(x, xb, n);
    }
    // 2) W -> bf16 + inv norms
    prep_w_kernel<<<HDIM, 256, 0, stream>>>(W, wb, invn);
    // 3) pre = x * W^T + b
    {
        dim3 grid(HDIM / 256, BDIM / 128);   // (48, 128)
        gemm_pre_kernel<<<grid, 256, 0, stream>>>(xb, wb, b, pre);
    }
    // 4) top-K threshold + sparse decode
    topk_decode_kernel<<<BDIM, 256, 0, stream>>>(pre, W, invn, out);
}
